// SudokuFactorizedAttentionBlock_91199335563373
// MI455X (gfx1250) — matrix operand order repaired and checked by reading, not compiled, not run on hardware
//
#include <hip/hip_runtime.h>
#include <math.h>

// ---------------------------------------------------------------------------
// Types
// ---------------------------------------------------------------------------
typedef __attribute__((ext_vector_type(16))) __bf16 v16bf;
typedef __attribute__((ext_vector_type(8)))  float  v8f;
typedef __attribute__((ext_vector_type(4)))  unsigned int u32x4;
typedef __attribute__((ext_vector_type(4)))  float  f32x4;

// Problem constants
#define BATCH 1024
#define LCELLS 81            // 9x9
#define DMODEL 512
#define HEADS 8
#define EDIM 64
#define FDIM 2048
#define MROWS (BATCH * LCELLS)   // 82944

// ---------------------------------------------------------------------------
// Weight transpose + f32 -> bf16 downconvert:  in[K,N] (row major) -> out[N,K]
// ---------------------------------------------------------------------------
__global__ void sudoku_transpose_w_bf16(const float* __restrict__ in,
                                        __bf16* __restrict__ out,
                                        int K, int N) {
    int total = K * N;
    for (int i = blockIdx.x * blockDim.x + threadIdx.x; i < total;
         i += gridDim.x * blockDim.x) {
        int n = i / K;
        int k = i - n * K;
        out[i] = (__bf16)in[k * N + n];   // coalesced writes along k
    }
}

// ---------------------------------------------------------------------------
// LayerNorm over D=512, one wave per row, bf16 output
// ---------------------------------------------------------------------------
__global__ void sudoku_ln_bf16(const float* __restrict__ in,
                               const float* __restrict__ gam,
                               const float* __restrict__ bet,
                               __bf16* __restrict__ out) {
    int wave = threadIdx.x >> 5;
    int lane = threadIdx.x & 31;
    size_t row = (size_t)blockIdx.x * 8 + wave;

    const f32x4* p = (const f32x4*)(in + row * DMODEL);
    f32x4 xv[4];
    float s1 = 0.f, s2 = 0.f;
#pragma unroll
    for (int i = 0; i < 4; ++i) {
        xv[i] = p[lane + 32 * i];
#pragma unroll
        for (int c = 0; c < 4; ++c) { float f = xv[i][c]; s1 += f; s2 += f * f; }
    }
#pragma unroll
    for (int o = 16; o > 0; o >>= 1) {
        s1 += __shfl_xor(s1, o, 32);
        s2 += __shfl_xor(s2, o, 32);
    }
    float mean = s1 * (1.f / DMODEL);
    float var  = s2 * (1.f / DMODEL) - mean * mean;
    float rstd = rsqrtf(var + 1e-5f);

    const f32x4* gp = (const f32x4*)gam;
    const f32x4* bp = (const f32x4*)bet;
    __bf16* orow = out + row * DMODEL;
#pragma unroll
    for (int i = 0; i < 4; ++i) {
        f32x4 gv = gp[lane + 32 * i];
        f32x4 bv = bp[lane + 32 * i];
        int base = (lane + 32 * i) * 4;
#pragma unroll
        for (int c = 0; c < 4; ++c)
            orow[base + c] = (__bf16)((xv[i][c] - mean) * rstd * gv[c] + bv[c]);
    }
}

// ---------------------------------------------------------------------------
// bf16 WMMA GEMM:  C[M,N] = A[M,K] @ W^T  (+bias, opt GELU, opt residual)
// A bf16 row-major (K contig); Bw bf16 [N,K] (K contig).
// Tile 128x128x32, 256 threads = 8 waves (2 x 4), 4x2 wmma accs per wave.
// Double-buffered LDS: one barrier per K-step.
// ---------------------------------------------------------------------------
__device__ inline float sudoku_gelu(float x) {
    return 0.5f * x * (1.0f + erff(x * 0.70710678118654752f));
}

#define LDSPAD 40   // row stride in bf16 units (20 dwords, conflict-free)

template <bool GELU, bool RES, bool OUTF>
__global__ __launch_bounds__(256)
void sudoku_gemm_bf16(const __bf16* __restrict__ A,
                      const __bf16* __restrict__ Bw,
                      const float* __restrict__ bias,
                      const float* __restrict__ res,
                      float* __restrict__ outF,
                      __bf16* __restrict__ outH,
                      int N, int K) {
    __shared__ __attribute__((aligned(16))) __bf16 As[2][128 * LDSPAD];
    __shared__ __attribute__((aligned(16))) __bf16 Bs[2][128 * LDSPAD];

    const int tid = threadIdx.x;
    const int rowBase = blockIdx.y * 128;
    const int colBase = blockIdx.x * 128;

    // global->LDS copy mapping: 8 bf16 (16B) per thread per half-tile
    const int lr = tid >> 2;            // 0..63
    const int lk = (tid & 3) << 3;      // 0,8,16,24 (bf16 units)

    const __bf16* Abase = A  + (size_t)rowBase * K;
    const __bf16* Bbase = Bw + (size_t)colBase * K;

    const int lane = tid & 31;
    const int wave = tid >> 5;
    const int wrow = wave & 1;          // 0..1 -> 64-row strip
    const int wcol = wave >> 1;         // 0..3 -> 32-col strip
    const int lm   = lane & 15;
    const int hi   = lane >> 4;         // 0/1

    v8f acc[4][2];
#pragma unroll
    for (int mt = 0; mt < 4; ++mt)
#pragma unroll
        for (int nt = 0; nt < 2; ++nt)
            acc[mt][nt] = (v8f){0.f, 0.f, 0.f, 0.f, 0.f, 0.f, 0.f, 0.f};

    union Frag { u32x4 u[2]; v16bf v; };

    auto fetch = [&](int k0, u32x4& a0, u32x4& a1, u32x4& b0, u32x4& b1) {
        a0 = *(const u32x4*)(Abase + (size_t)lr * K + k0 + lk);
        a1 = *(const u32x4*)(Abase + (size_t)(lr + 64) * K + k0 + lk);
        b0 = *(const u32x4*)(Bbase + (size_t)lr * K + k0 + lk);
        b1 = *(const u32x4*)(Bbase + (size_t)(lr + 64) * K + k0 + lk);
    };
    auto deposit = [&](int buf, u32x4 a0, u32x4 a1, u32x4 b0, u32x4 b1) {
        *(u32x4*)(As[buf] + lr * LDSPAD + lk) = a0;
        *(u32x4*)(As[buf] + (lr + 64) * LDSPAD + lk) = a1;
        *(u32x4*)(Bs[buf] + lr * LDSPAD + lk) = b0;
        *(u32x4*)(Bs[buf] + (lr + 64) * LDSPAD + lk) = b1;
    };
    auto compute = [&](int buf) {
        // A frags: 16x32 bf16; lane m = lm, regs v0..3 <- K{koff..+7},
        // v4..7 <- K{16+koff..}, koff = hi*8
        v16bf af[4];
#pragma unroll
        for (int mt = 0; mt < 4; ++mt) {
            Frag f;
            const __bf16* rp = As[buf] + (wrow * 64 + mt * 16 + lm) * LDSPAD + hi * 8;
            f.u[0] = *(const u32x4*)(rp);
            f.u[1] = *(const u32x4*)(rp + 16);
            af[mt] = f.v;
        }
        // B frags: 32x16 bf16; lane n = lm, regs v0..3 <- K{koff..+7}, v4..7 <- K{16+koff..}, koff = hi*8
        v16bf bfr[2];
#pragma unroll
        for (int nt = 0; nt < 2; ++nt) {
            Frag f;
            const __bf16* rp = Bs[buf] + (wcol * 32 + nt * 16 + lm) * LDSPAD + hi * 8;
            f.u[0] = *(const u32x4*)(rp);
            f.u[1] = *(const u32x4*)(rp + 16);
            bfr[nt] = f.v;
        }
#pragma unroll
        for (int mt = 0; mt < 4; ++mt)
#pragma unroll
            for (int nt = 0; nt < 2; ++nt)
                acc[mt][nt] = __builtin_amdgcn_wmma_f32_16x16x32_bf16(
                    false, af[mt], false, bfr[nt], (short)0, acc[mt][nt],
                    false, false);
    };

    const int KT = K >> 5;

    {
        u32x4 a0, a1, b0, b1;
        fetch(0, a0, a1, b0, b1);
        deposit(0, a0, a1, b0, b1);
    }
    __syncthreads();

    int cur = 0;
    for (int kt = 0; kt < KT - 1; ++kt) {
        u32x4 a0, a1, b0, b1;
        fetch((kt + 1) << 5, a0, a1, b0, b1);   // prefetch next tile
        compute(cur);                            // 8 WMMAs from buf cur
        deposit(cur ^ 1, a0, a1, b0, b1);        // fill other buffer
        __syncthreads();                         // single barrier per step
        cur ^= 1;
    }
    compute(cur);                                // last K-step

    // Epilogue: C layout -> lane l: col = lm; reg i: row = i (+8 if hi)
#pragma unroll
    for (int nt = 0; nt < 2; ++nt) {
        int col = colBase + wcol * 32 + nt * 16 + lm;
        float bv = bias[col];
#pragma unroll
        for (int mt = 0; mt < 4; ++mt) {
            size_t base = (size_t)(rowBase + wrow * 64 + mt * 16 + hi * 8) * N + col;
#pragma unroll
            for (int i = 0; i < 8; ++i) {
                float vv = acc[mt][nt][i] + bv;
                if (GELU) vv = sudoku_gelu(vv);
                size_t off = base + (size_t)i * N;
                if (RES)  vv += res[off];
                if (OUTF) outF[off] = vv;
                else      outH[off] = (__bf16)vv;
            }
        }
    }
}

// ---------------------------------------------------------------------------
// Factorized 9-way attention (row, col, 3x3 block), averaged.
// One block per (batch, head). qkv bf16 [M,1536]; y bf16 [M,512].
// ---------------------------------------------------------------------------
__global__ __launch_bounds__(256)
void sudoku_attn(const __bf16* __restrict__ qkv, __bf16* __restrict__ y) {
    __shared__ float qs[LCELLS * EDIM];       // [t][e]
    __shared__ float vs[LCELLS * EDIM];       // [t][e]
    __shared__ float kT[EDIM * LCELLS];       // [e][t]  (stride 81: odd, no conflicts)
    __shared__ float scr[8][32];
    __shared__ int   icr[8][32];

    int bh = blockIdx.x;
    int b = bh >> 3;
    int h = bh & 7;
    const __bf16* base = qkv + (size_t)b * LCELLS * (3 * DMODEL) + h * EDIM;

    for (int idx = threadIdx.x; idx < LCELLS * EDIM; idx += 256) {
        int t = idx >> 6;
        int e = idx & 63;
        size_t r = (size_t)t * (3 * DMODEL) + e;
        qs[idx]            = (float)base[r];
        kT[e * LCELLS + t] = (float)base[r + DMODEL];
        vs[idx]            = (float)base[r + 2 * DMODEL];
    }
    __syncthreads();

    int wave = threadIdx.x >> 5;
    int lane = threadIdx.x & 31;
    volatile float* sw = scr[wave];
    volatile int*   iw = icr[wave];

    for (int t = wave; t < LCELLS; t += 8) {
        int r = t / 9, c = t - r * 9;
        int cell;
        if (lane < 9)        cell = r * 9 + lane;
        else if (lane < 18)  cell = (lane - 9) * 9 + c;
        else if (lane < 27) { int i = lane - 18;
                              cell = ((r / 3) * 3 + i / 3) * 9 + (c / 3) * 3 + (i % 3); }
        else                 cell = t;

        // score: q[t] . k[cell]  (q broadcast, kT lanes spread over cells)
        const float* qt = qs + t * EDIM;
        float s = 0.f;
#pragma unroll 8
        for (int e = 0; e < EDIM; ++e) s += qt[e] * kT[e * LCELLS + cell];
        s *= 0.125f;                       // 1/sqrt(64)

        sw[lane] = s;
        iw[lane] = cell;

        // softmax within each group of 9 lanes (wave-lockstep through LDS)
        int g0 = (lane < 27) ? (lane / 9) * 9 : 0;
        float m = -INFINITY;
#pragma unroll
        for (int i = 0; i < 9; ++i) m = fmaxf(m, sw[g0 + i]);
        float p = __expf(s - m);
        sw[lane] = (lane < 27) ? p : 0.f;
        float den = 0.f;
#pragma unroll
        for (int i = 0; i < 9; ++i) den += sw[g0 + i];
        sw[lane] = p / den;

        // weighted sum over the 27 (pattern, key) pairs; all lanes, 2 dims each
        float a0 = 0.f, a1 = 0.f;
#pragma unroll 9
        for (int j = 0; j < 27; ++j) {
            float pj = sw[j];
            const float* vr = vs + iw[j] * EDIM;
            a0 += pj * vr[lane];
            a1 += pj * vr[lane + 32];
        }
        size_t o = ((size_t)(b * LCELLS + t)) * DMODEL + h * EDIM;
        y[o + lane]      = (__bf16)(a0 * (1.f / 3.f));
        y[o + lane + 32] = (__bf16)(a1 * (1.f / 3.f));
    }
}

// ---------------------------------------------------------------------------
// Host launcher
// ---------------------------------------------------------------------------
extern "C" void kernel_launch(void* const* d_in, const int* in_sizes, int n_in,
                              void* d_out, int out_size, void* d_ws, size_t ws_size,
                              hipStream_t stream) {
    const float* src    = (const float*)d_in[0];
    const float* w_qkv  = (const float*)d_in[1];
    const float* b_qkv  = (const float*)d_in[2];
    const float* w_out  = (const float*)d_in[3];
    const float* b_out  = (const float*)d_in[4];
    const float* g1     = (const float*)d_in[5];
    const float* beta1  = (const float*)d_in[6];
    const float* g2     = (const float*)d_in[7];
    const float* beta2  = (const float*)d_in[8];
    const float* w_ff1  = (const float*)d_in[9];
    const float* b_ff1  = (const float*)d_in[10];
    const float* w_ff2  = (const float*)d_in[11];
    const float* b_ff2  = (const float*)d_in[12];

    const size_t M = MROWS;                       // 82944
    char* ws = (char*)d_ws;

    const size_t SZ_XLN = M * DMODEL * 2;         // 84,934,656  (bf16)
    const size_t SZ_Y   = M * DMODEL * 2;
    const size_t SZ_X   = M * DMODEL * 4;         // 169,869,312 (f32)
    const size_t SZ_BIG = M * FDIM * 2;           // 339,738,624 (>= qkv bf16)

    __bf16* xln  = (__bf16*)(ws);                          // ln1 out, later ln2 out
    __bf16* ybuf = (__bf16*)(ws + SZ_XLN);                 // attention out
    float*  xres = (float*)(ws + SZ_XLN + SZ_Y);           // x = src + attn-proj
    char*   big  =          ws + SZ_XLN + SZ_Y + SZ_X;
    __bf16* qkvb = (__bf16*)big;                           // qkv bf16 [M,1536]
    __bf16* hmid = (__bf16*)big;                           // reused: gelu(ff1) [M,2048]
    __bf16* wqkvT = (__bf16*)(big + SZ_BIG);               // [1536,512]
    __bf16* woutT = wqkvT + (size_t)(3 * DMODEL) * DMODEL; // [512,512]
    __bf16* wff1T = woutT + (size_t)DMODEL * DMODEL;       // [2048,512]
    __bf16* wff2T = wff1T + (size_t)FDIM * DMODEL;         // [512,2048]

    // 1) weight transposes (K,N) -> bf16 [N,K]
    sudoku_transpose_w_bf16<<<3072, 256, 0, stream>>>(w_qkv, wqkvT, DMODEL, 3 * DMODEL);
    sudoku_transpose_w_bf16<<<1024, 256, 0, stream>>>(w_out, woutT, DMODEL, DMODEL);
    sudoku_transpose_w_bf16<<<4096, 256, 0, stream>>>(w_ff1, wff1T, DMODEL, FDIM);
    sudoku_transpose_w_bf16<<<4096, 256, 0, stream>>>(w_ff2, wff2T, FDIM, DMODEL);

    // 2) ln1(src) -> xln (bf16)
    sudoku_ln_bf16<<<M / 8, 256, 0, stream>>>(src, g1, beta1, xln);

    // 3) qkv = xln @ w_qkv + b_qkv   -> bf16 [M,1536]
    sudoku_gemm_bf16<false, false, false>
        <<<dim3((3 * DMODEL) / 128, M / 128), 256, 0, stream>>>(
        xln, wqkvT, b_qkv, nullptr, nullptr, qkvb, 3 * DMODEL, DMODEL);

    // 4) factorized attention -> ybuf (bf16 [M,512])
    sudoku_attn<<<BATCH * HEADS, 256, 0, stream>>>(qkvb, ybuf);

    // 5) x = src + ybuf @ w_out + b_out  -> xres (f32)
    sudoku_gemm_bf16<false, true, true>
        <<<dim3(DMODEL / 128, M / 128), 256, 0, stream>>>(
        ybuf, woutT, b_out, src, xres, nullptr, DMODEL, DMODEL);

    // 6) ln2(x) -> xln (reuse, bf16)
    sudoku_ln_bf16<<<M / 8, 256, 0, stream>>>(xres, g2, beta2, xln);

    // 7) hmid = gelu(xln @ w_ff1 + b_ff1)  -> bf16 [M,2048]
    sudoku_gemm_bf16<true, false, false>
        <<<dim3(FDIM / 128, M / 128), 256, 0, stream>>>(
        xln, wff1T, b_ff1, nullptr, nullptr, hmid, FDIM, DMODEL);

    // 8) out = x + hmid @ w_ff2 + b_ff2   -> f32 d_out
    sudoku_gemm_bf16<false, true, true>
        <<<dim3(DMODEL / 128, M / 128), 256, 0, stream>>>(
        hmid, wff2T, b_ff2, xres, (float*)d_out, nullptr, DMODEL, FDIM);
}
